// PointTransformerBatchLayer_5617817224087
// MI455X (gfx1250) — matrix operand order repaired and checked by reading, not compiled, not run on hardware
//
#include <hip/hip_runtime.h>
#include <cstddef>

typedef _Float16 h16;
typedef __attribute__((ext_vector_type(16))) _Float16 v16h;
typedef __attribute__((ext_vector_type(8)))  float    v8f;

// ---------------- problem constants ----------------
#define NPTS   1024
#define DIMCH  64
#define KNB    16
#define NANCH  12
#define HID    256     // ATTN_HID
#define TILE_P 8
#define NCOL   (TILE_P * KNB)   // 128 columns per block

// LDS strides (padded to dodge bank conflicts)
#define LDX   72      // halfs  (64 + 8)
#define LDH   264     // halfs  (256 + 8)
#define LDPE  68      // floats (64 + 4)
#define LDSS  136     // floats (128 + 8)

#define SZ_PE  (NCOL * LDPE * 4)          // 34816
#define SZ_B   (NCOL * LDH * 2)           // 67584 (holds pe1 then H)
#define SZ_C   (DIMCH * LDSS * 4)         // 34816 (holds X then S)
#define SZ_GX  (3 * NCOL * 4)             // 1536
#define SZ_IDX (NCOL * 4)                 // 512
#define SMEM_BYTES (SZ_PE + SZ_B + SZ_C + SZ_GX + SZ_IDX)   // 139264 B

// ---------------- WMMA fragment helpers (wave32 layouts per CDNA5 ISA 7.12.2) ----------------
// A (16x32 f16, row-major weights, K contiguous): lane L -> M = m0 + L%16,
// K offset 8*(L/16); VGPR j holds K = (j&3)*2 + (j>>2)*16 (+base), pairs contiguous.
__device__ __forceinline__ v16h frag_a(const h16* __restrict__ W, int ld, int m0, int k0, int lane) {
  const h16* row = W + (m0 + (lane & 15)) * ld + k0 + ((lane >> 4) << 3);
  v16h f;
#pragma unroll
  for (int j = 0; j < 8; ++j) {
    int k = ((j & 3) << 1) + ((j >> 2) << 4);
    f[2 * j]     = row[k];
    f[2 * j + 1] = row[k + 1];
  }
  return f;
}

// B (32x16 f16) from LDS stored column-major [col][K] (K contiguous per column):
// lane L -> col = n0 + L%16, K base = k0 + 8*(L/16); elements 0..7 -> K base+0..7, 8..15 -> K base+16..23.
__device__ __forceinline__ v16h frag_b(const h16* __restrict__ X, int ld, int k0, int n0, int lane) {
  const h16* p = X + (n0 + (lane & 15)) * ld + k0 + ((lane >> 4) << 3);
  v16h f;
#pragma unroll
  for (int j = 0; j < 16; ++j) f[j] = p[j + ((j >> 3) << 3)];
  return f;
}

__device__ __forceinline__ v8f wmma32(v16h a, v16h b, v8f c) {
  return __builtin_amdgcn_wmma_f32_16x16x32_f16(false, a, false, b, (short)0, c, false, false);
}

// ---------------- kernel 1: f32 -> f16 weight conversion ----------------
__global__ __launch_bounds__(256) void prep_kernel(const float* __restrict__ w1,
                                                   const float* __restrict__ w2,
                                                   const float* __restrict__ p2,
                                                   h16* __restrict__ w1h,
                                                   h16* __restrict__ w2h,
                                                   h16* __restrict__ p2h) {
  int i = blockIdx.x * 256 + threadIdx.x;      // 64 blocks -> 16384 threads
  if (i < HID * DIMCH) w1h[i] = (h16)w1[i];
  if (i < DIMCH * HID) w2h[i] = (h16)w2[i];
  if (i < DIMCH * DIMCH) p2h[i] = (h16)p2[i];
}

// ---------------- kernel 2: exact KNN (top-16 by squared distance, self included) ----------------
__global__ __launch_bounds__(256) void knn_kernel(const float* __restrict__ xyz,
                                                  int* __restrict__ knn_idx) {
  __shared__ float sx[NPTS], sy[NPTS], sz[NPTS];
  for (int i = threadIdx.x; i < NPTS; i += 256) {
    sx[i] = xyz[i];
    sy[i] = xyz[NPTS + i];
    sz[i] = xyz[2 * NPTS + i];
  }
  __syncthreads();

  int n = blockIdx.x * 256 + threadIdx.x;      // 4 blocks -> 1024 queries
  float px = sx[n], py = sy[n], pz = sz[n];

  float bd[KNB];
  int   bi[KNB];
#pragma unroll
  for (int i = 0; i < KNB; ++i) { bd[i] = 3.4e38f; bi[i] = 0; }

  for (int j = 0; j < NPTS; ++j) {
    float dx = px - sx[j], dy = py - sy[j], dz = pz - sz[j];
    float dd = dx * dx + dy * dy + dz * dz;
    int jj = j;
#pragma unroll
    for (int i = 0; i < KNB; ++i) {            // register bubble-insert, stable on ties
      bool sw = dd < bd[i];
      float td = bd[i]; int ti = bi[i];
      if (sw) { bd[i] = dd; bi[i] = jj; dd = td; jj = ti; }
    }
  }
#pragma unroll
  for (int i = 0; i < KNB; ++i) knn_idx[n * KNB + i] = bi[i];
}

// ---------------- kernel 3: qkv projection into workspace, layout [a][n][192] ----------------
__global__ __launch_bounds__(256) void qkv_kernel(const float* __restrict__ to_qkv,
                                                  const float* __restrict__ feats,
                                                  float* __restrict__ qkv) {
  int gid = blockIdx.x * 256 + threadIdx.x;    // 192 * 12288 outputs
  int j   = gid % (3 * DIMCH);
  int col = gid / (3 * DIMCH);
  int n   = col & (NPTS - 1);
  int a   = col >> 10;
  const float* w = to_qkv + j * DIMCH;
  float acc = 0.f;
#pragma unroll
  for (int c = 0; c < DIMCH; ++c)
    acc += w[c] * feats[(c * NPTS + n) * NANCH + a];
  qkv[((size_t)(a * NPTS + n)) * (3 * DIMCH) + j] = acc;
}

// ---------------- kernel 4: fused pe-MLP + attn-MLP + softmax + aggregate ----------------
// grid = (N/TILE_P, NA); block = 256 threads = 8 waves.
__global__ __launch_bounds__(256) void fused_kernel(const float* __restrict__ xyz,
                                                    const float* __restrict__ anchors,
                                                    const float* __restrict__ pos_mlp1,
                                                    const float* __restrict__ qkv,
                                                    const int* __restrict__ knn_idx,
                                                    const h16* __restrict__ P2h,
                                                    const h16* __restrict__ W1h,
                                                    const h16* __restrict__ W2h,
                                                    float* __restrict__ out) {
  __shared__ __align__(16) unsigned char smem[SMEM_BYTES];
  float* s_pe  = (float*)(smem);                               // [NCOL][LDPE] f32
  h16*   s_pe1 = (h16*)(smem + SZ_PE);                         // [NCOL][LDX]  (phase A)
  h16*   s_H   = (h16*)(smem + SZ_PE);                         // [NCOL][LDH]  (phase C/D)
  h16*   s_X   = (h16*)(smem + SZ_PE + SZ_B);                  // [NCOL][LDX]  (phase B/C)
  float* s_S   = (float*)(smem + SZ_PE + SZ_B);                // [DIMCH][LDSS] (phase D/E)
  float* s_gx  = (float*)(smem + SZ_PE + SZ_B + SZ_C);         // [3][NCOL]
  int*   s_idx = (int*)(smem + SZ_PE + SZ_B + SZ_C + SZ_GX);   // [NCOL]

  const int tid  = threadIdx.x;
  const int lane = tid & 31;
  const int wv   = tid >> 5;
  const int p0   = blockIdx.x * TILE_P;
  const int a    = blockIdx.y;
  const float* qbase = qkv + (size_t)a * NPTS * (3 * DIMCH);

  // ---- weight prefetch: 128B cachelines covering W1 (32KB), W2 (32KB), P2 (8KB) ----
  __builtin_prefetch((const char*)W1h + tid * 128, 0, 3);
  __builtin_prefetch((const char*)W2h + tid * 128, 0, 3);
  if (tid < 64) __builtin_prefetch((const char*)P2h + tid * 128, 0, 3);

  // ---- phase 0: neighbor indices + anchor-rotated relative positions ----
  if (tid < NCOL) {
    int n = p0 + (tid >> 4);
    int m = knn_idx[n * KNB + (tid & 15)];
    s_idx[tid] = m;
    float rx = xyz[n] - xyz[m];
    float ry = xyz[NPTS + n] - xyz[NPTS + m];
    float rz = xyz[2 * NPTS + n] - xyz[2 * NPTS + m];
    const float* R = anchors + a * 9;
    s_gx[0 * NCOL + tid] = R[0] * rx + R[1] * ry + R[2] * rz;
    s_gx[1 * NCOL + tid] = R[3] * rx + R[4] * ry + R[5] * rz;
    s_gx[2 * NCOL + tid] = R[6] * rx + R[7] * ry + R[8] * rz;
  }
  __syncthreads();

  // ---- phase A: pe1 = relu(pos_mlp1 @ gx)  (K=3, plain VALU) ----
  for (int e = tid; e < NCOL * DIMCH; e += 256) {
    int col = e >> 6, h = e & 63;
    float g0 = s_gx[col], g1 = s_gx[NCOL + col], g2 = s_gx[2 * NCOL + col];
    const float* p1 = pos_mlp1 + h * 3;
    float v = p1[0] * g0 + p1[1] * g1 + p1[2] * g2;
    s_pe1[col * LDX + h] = (h16)fmaxf(v, 0.f);
  }
  __syncthreads();

  // ---- phase A3: pe = pos_mlp2 @ pe1  (WMMA; per-wave mt is fixed, hoist A frags) ----
  {
    const int mt = wv >> 1;                       // 4 M-tiles over 8 waves (2 waves each)
    const v16h pa0 = frag_a(P2h, DIMCH, mt * 16, 0,  lane);
    const v16h pa1 = frag_a(P2h, DIMCH, mt * 16, 32, lane);
#pragma unroll
    for (int q = 0; q < 4; ++q) {
      int nt = ((wv & 1) << 2) + q;
      v8f acc = {};
      acc = wmma32(pa0, frag_b(s_pe1, LDX, 0,  nt * 16, lane), acc);
      acc = wmma32(pa1, frag_b(s_pe1, LDX, 32, nt * 16, lane), acc);
      int c0 = nt * 16 + (lane & 15);
#pragma unroll
      for (int j = 0; j < 8; ++j)
        s_pe[c0 * LDPE + mt * 16 + j + ((lane >> 4) << 3)] = acc[j];
    }
  }
  __syncthreads();

  // ---- phase B: X = q - k_gathered + pe  -> f16, column-major [col][ch], float4 path ----
  for (int e = tid; e < NCOL * (DIMCH / 4); e += 256) {   // 2048 quad-items
    int col = e >> 4, c4 = (e & 15) << 2;
    int n = p0 + (col >> 4);
    int m = s_idx[col];
    const float4 qv = *(const float4*)(qbase + (size_t)n * 192 + c4);
    const float4 kv = *(const float4*)(qbase + (size_t)m * 192 + 64 + c4);
    const float4 pv = *(const float4*)(s_pe + col * LDPE + c4);
    h16* xp = s_X + col * LDX + c4;
    xp[0] = (h16)(qv.x - kv.x + pv.x);
    xp[1] = (h16)(qv.y - kv.y + pv.y);
    xp[2] = (h16)(qv.z - kv.z + pv.z);
    xp[3] = (h16)(qv.w - kv.w + pv.w);
  }
  __syncthreads();

  // ---- phase C: H = relu(attn_mlp1 @ X)  (WMMA; 2 mt per wave, A frags hoisted) ----
#pragma unroll
  for (int mi = 0; mi < 2; ++mi) {
    const int mt = (wv << 1) + mi;                // 16 M-tiles over 8 waves
    const v16h wa0 = frag_a(W1h, DIMCH, mt * 16, 0,  lane);
    const v16h wa1 = frag_a(W1h, DIMCH, mt * 16, 32, lane);
#pragma unroll
    for (int nt = 0; nt < 8; ++nt) {
      v8f acc = {};
      acc = wmma32(wa0, frag_b(s_X, LDX, 0,  nt * 16, lane), acc);
      acc = wmma32(wa1, frag_b(s_X, LDX, 32, nt * 16, lane), acc);
      int c0 = nt * 16 + (lane & 15);
#pragma unroll
      for (int j = 0; j < 8; ++j)
        s_H[c0 * LDH + mt * 16 + j + ((lane >> 4) << 3)] = (h16)fmaxf(acc[j], 0.f);
    }
  }
  __syncthreads();

  // ---- phase D: S = attn_mlp2 @ H  (WMMA; per-wave mt fixed, 8 K-step A frags hoisted) ----
  {
    const int mt = wv >> 1;                       // 4 M-tiles over 8 waves
    v16h aw[8];
#pragma unroll
    for (int ks = 0; ks < 8; ++ks)
      aw[ks] = frag_a(W2h, HID, mt * 16, ks * 32, lane);
#pragma unroll
    for (int q = 0; q < 4; ++q) {
      int nt = ((wv & 1) << 2) + q;
      v8f acc = {};
#pragma unroll
      for (int ks = 0; ks < 8; ++ks)
        acc = wmma32(aw[ks], frag_b(s_H, LDH, ks * 32, nt * 16, lane), acc);
      int c0 = nt * 16 + (lane & 15);
#pragma unroll
      for (int j = 0; j < 8; ++j)
        s_S[(mt * 16 + j + ((lane >> 4) << 3)) * LDSS + c0] = acc[j];
    }
  }
  __syncthreads();

  // ---- phase E: per-(channel, point) softmax over 16 neighbors + aggregate ----
  for (int e = tid; e < DIMCH * TILE_P; e += 256) {   // 512 pairs
    int c = e & 63, p = e >> 6;
    float sv[KNB];
    float mx = -3.4e38f;
#pragma unroll
    for (int k = 0; k < KNB; ++k) {
      sv[k] = s_S[c * LDSS + p * 16 + k];
      mx = fmaxf(mx, sv[k]);
    }
    float sum = 0.f;
#pragma unroll
    for (int k = 0; k < KNB; ++k) { sv[k] = __expf(sv[k] - mx); sum += sv[k]; }
    float inv = 1.f / sum;
    float accv = 0.f;
#pragma unroll
    for (int k = 0; k < KNB; ++k) {
      int col = p * 16 + k;
      int m = s_idx[col];
      float vg = qbase[(size_t)m * 192 + 128 + c] + s_pe[col * LDPE + c];
      accv += sv[k] * inv * vg;
    }
    int n = p0 + p;
    out[((size_t)c * NPTS + n) * NANCH + a] = accv;
  }
}

// ---------------- launcher ----------------
extern "C" void kernel_launch(void* const* d_in, const int* in_sizes, int n_in,
                              void* d_out, int out_size, void* d_ws, size_t ws_size,
                              hipStream_t stream) {
  const float* xyz       = (const float*)d_in[0];
  const float* feats     = (const float*)d_in[1];
  const float* anchors   = (const float*)d_in[2];
  const float* to_qkv    = (const float*)d_in[3];
  const float* pos_mlp1  = (const float*)d_in[4];
  const float* pos_mlp2  = (const float*)d_in[5];
  const float* attn_mlp1 = (const float*)d_in[6];
  const float* attn_mlp2 = (const float*)d_in[7];
  float* out = (float*)d_out;

  unsigned char* ws = (unsigned char*)d_ws;
  int*   knn_idx = (int*)ws;                               // 1024*16*4      = 65536 B
  float* qkvb    = (float*)(ws + 65536);                   // 12*1024*192*4  = 9437184 B
  h16*   w1h     = (h16*)(ws + 65536 + 9437184);           // 256*64*2
  h16*   w2h     = w1h + HID * DIMCH;                      // 64*256*2
  h16*   p2h     = w2h + DIMCH * HID;                      // 64*64*2

  prep_kernel<<<dim3(64), dim3(256), 0, stream>>>(attn_mlp1, attn_mlp2, pos_mlp2,
                                                  w1h, w2h, p2h);
  knn_kernel<<<dim3(NPTS / 256), dim3(256), 0, stream>>>(xyz, knn_idx);
  qkv_kernel<<<dim3((3 * DIMCH * NPTS * NANCH) / 256), dim3(256), 0, stream>>>(to_qkv, feats, qkvb);
  fused_kernel<<<dim3(NPTS / TILE_P, NANCH), dim3(256), 0, stream>>>(
      xyz, anchors, pos_mlp1, qkvb, knn_idx, p2h, w1h, w2h, out);
}
